// MultiDilatelocalAttention_26491358282247
// MI455X (gfx1250) — matrix operand order repaired and checked by reading, not compiled, not run on hardware
//
#include <hip/hip_runtime.h>
#include <hip/hip_bf16.h>

// ---------------------------------------------------------------------------
// Multi-dilate local attention, MI455X (gfx1250, wave32, WMMA).
// Stage 1: qkv = x @ Wqkv^T          (bf16 WMMA, f32 accum) -> ws
// Stage 2: dilated 3x3 window attention (dil 2 and 3)       -> ws
// Stage 3: out = xo @ Wproj^T + b    (bf16 WMMA, f32 accum) -> d_out
// Memory-bound (~0.7 GB @ 23.3 TB/s); GEMMs use double-buffered LDS tiles
// with register staging so HBM/L2 latency overlaps the WMMA stream.
// ---------------------------------------------------------------------------

typedef __attribute__((ext_vector_type(16))) __bf16 v16bf;
typedef __attribute__((ext_vector_type(8)))  __bf16 v8bf;
typedef __attribute__((ext_vector_type(8)))  float  v8f;

union FragBF { v16bf v; v8bf h[2]; };

// f32 -> bf16 (round-to-nearest-even), raw bits (fallback path)
__device__ __forceinline__ unsigned short f2bf_bits(float f) {
    unsigned u = __builtin_bit_cast(unsigned, f);
    u += 0x7FFFu + ((u >> 16) & 1u);
    return (unsigned short)(u >> 16);
}

#if __has_builtin(__builtin_amdgcn_cvt_pk_bf16_f32)
// hardware packed converter: v_cvt_pk_bf16_f32 (2 f32 -> 1 dword of bf16x2)
__device__ __forceinline__ unsigned pack2bf(float x, float y) {
    auto r = __builtin_amdgcn_cvt_pk_bf16_f32(x, y);
    unsigned u;
    __builtin_memcpy(&u, &r, 4);
    return u;
}
#else
__device__ __forceinline__ unsigned pack2bf(float x, float y) {
    return (unsigned)f2bf_bits(x) | ((unsigned)f2bf_bits(y) << 16);
}
#endif

__device__ __forceinline__ unsigned long long pack4bf(float4 d) {
    return (unsigned long long)pack2bf(d.x, d.y) |
           ((unsigned long long)pack2bf(d.z, d.w) << 32);
}

// ---------------------------------------------------------------------------
// GEMM: Cout[M,N] = A[M,K] * W[N,K]^T (+ bias[N]); all f32 row-major.
// Block = 256 thr (8 waves). Block tile 128(M) x 64(N). K stepped by 32,
// fully unrolled (K compile-time). Double-buffered LDS, register-staged
// prefetch of the next K-tile.
// ---------------------------------------------------------------------------
#define LDS_STRIDE 48  // 32 payload + 16 pad halves; 96B rows keep 16B frags aligned

template <int K>
__global__ __launch_bounds__(256)
void gemm_bf16_wmma(const float* __restrict__ A, const float* __restrict__ W,
                    float* __restrict__ Cout, const float* __restrict__ bias,
                    int M, int N) {
    __shared__ unsigned short As[2][128][LDS_STRIDE]; // [buf][m][k] bf16 bits
    __shared__ unsigned short Bs[2][64][LDS_STRIDE];  // [buf][n][k] bf16 bits

    const int t    = threadIdx.x;
    const int wave = t >> 5;
    const int lane = t & 31;
    const int mBase = blockIdx.y * 128;
    const int nBase = blockIdx.x * 64;

    const int mRow = wave * 16 + (lane & 15); // A-frag row within block tile
    const int kq   = lane >> 4;               // lane half selector

    v8f acc[4];
    const v8f vzero = {0.f, 0.f, 0.f, 0.f, 0.f, 0.f, 0.f, 0.f};
#pragma unroll
    for (int nt = 0; nt < 4; ++nt) acc[nt] = vzero;

    constexpr int KSTEPS = K >> 5;
    float4 ra[4], rb[2];

    // ---- prologue: fetch K-step 0 into registers, stage into buffer 0
#pragma unroll
    for (int i = 0; i < 4; ++i) {
        int idx = t + i * 256, row = idx >> 3, c4 = idx & 7;
        ra[i] = *reinterpret_cast<const float4*>(
            A + (size_t)(mBase + row) * K + c4 * 4);
    }
#pragma unroll
    for (int i = 0; i < 2; ++i) {
        int idx = t + i * 256, row = idx >> 3, c4 = idx & 7;
        rb[i] = *reinterpret_cast<const float4*>(
            W + (size_t)(nBase + row) * K + c4 * 4);
    }
#pragma unroll
    for (int i = 0; i < 4; ++i) {
        int idx = t + i * 256, row = idx >> 3, c4 = idx & 7;
        *reinterpret_cast<unsigned long long*>(&As[0][row][c4 * 4]) = pack4bf(ra[i]);
    }
#pragma unroll
    for (int i = 0; i < 2; ++i) {
        int idx = t + i * 256, row = idx >> 3, c4 = idx & 7;
        *reinterpret_cast<unsigned long long*>(&Bs[0][row][c4 * 4]) = pack4bf(rb[i]);
    }
    __syncthreads();

#pragma unroll
    for (int step = 0; step < KSTEPS; ++step) {
        const int cur = step & 1;

        // ---- issue next tile's global loads first (latency hidden by WMMAs)
        if (step + 1 < KSTEPS) {
            const int k0 = (step + 1) << 5;
#pragma unroll
            for (int i = 0; i < 4; ++i) {
                int idx = t + i * 256, row = idx >> 3, c4 = idx & 7;
                ra[i] = *reinterpret_cast<const float4*>(
                    A + (size_t)(mBase + row) * K + k0 + c4 * 4);
            }
#pragma unroll
            for (int i = 0; i < 2; ++i) {
                int idx = t + i * 256, row = idx >> 3, c4 = idx & 7;
                rb[i] = *reinterpret_cast<const float4*>(
                    W + (size_t)(nBase + row) * K + k0 + c4 * 4);
            }
        }

        // ---- A fragment (ISA 16-bit A 16x32 layout):
        // lanes 0-15: K 0-7 & 16-23 ; lanes 16-31: K 8-15 & 24-31
        FragBF a;
        a.h[0] = *reinterpret_cast<const v8bf*>(&As[cur][mRow][kq * 8]);
        a.h[1] = *reinterpret_cast<const v8bf*>(&As[cur][mRow][16 + kq * 8]);

        // ---- preload all 4 B fragments, then issue WMMAs back-to-back
        FragBF bf4[4];
#pragma unroll
        for (int nt = 0; nt < 4; ++nt) {
            const int nRow = nt * 16 + (lane & 15);
            bf4[nt].h[0] = *reinterpret_cast<const v8bf*>(&Bs[cur][nRow][kq * 8]);
            bf4[nt].h[1] = *reinterpret_cast<const v8bf*>(&Bs[cur][nRow][16 + kq * 8]);
        }
#pragma unroll
        for (int nt = 0; nt < 4; ++nt)
            acc[nt] = __builtin_amdgcn_wmma_f32_16x16x32_bf16(
                false, a.v, false, bf4[nt].v, (short)0, acc[nt], false, false);

        // ---- convert + stage next tile into alternate buffer
        if (step + 1 < KSTEPS) {
            const int nxt = cur ^ 1;
#pragma unroll
            for (int i = 0; i < 4; ++i) {
                int idx = t + i * 256, row = idx >> 3, c4 = idx & 7;
                *reinterpret_cast<unsigned long long*>(&As[nxt][row][c4 * 4]) =
                    pack4bf(ra[i]);
            }
#pragma unroll
            for (int i = 0; i < 2; ++i) {
                int idx = t + i * 256, row = idx >> 3, c4 = idx & 7;
                *reinterpret_cast<unsigned long long*>(&Bs[nxt][row][c4 * 4]) =
                    pack4bf(rb[i]);
            }
        }
        __syncthreads();
    }

    // ---- store: C/D frag lane l: M = r + 8*(l>>4), N = l&15 (coalesced rows)
#pragma unroll
    for (int nt = 0; nt < 4; ++nt) {
        const int n = nBase + nt * 16 + (lane & 15);
        const float bv = bias ? bias[n] : 0.f;
#pragma unroll
        for (int r = 0; r < 8; ++r) {
            const int m = mBase + wave * 16 + r + ((lane >> 4) << 3);
            Cout[(size_t)m * N + n] = acc[nt][r] + bv;
        }
    }
}

// ---------------------------------------------------------------------------
// Dilated local attention. qkv layout: [pixel(65536)][o(768)] with
// o = sel*256 + branch*128 + head*32 + d  (sel: 0=q,1=k,2=v).
// One wave per (b, pixel, branch, head); lane = d. Every q/k/v access is a
// single coalesced 128B line per wave. Zero-padded taps participate in the
// softmax with score 0 (matches the reference's zero padding).
// ---------------------------------------------------------------------------
__global__ __launch_bounds__(256)
void dilated_attn(const float* __restrict__ qkv, float* __restrict__ xo) {
    const int wid  = (int)((blockIdx.x * (unsigned)blockDim.x + threadIdx.x) >> 5);
    const int lane = threadIdx.x & 31;

    const int hw = wid & 16383;
    int tmp = wid >> 14;
    const int head   = tmp & 3;  tmp >>= 2;
    const int branch = tmp & 1;
    const int b      = tmp >> 1;

    const int h = hw >> 7, w = hw & 127;
    const int dil = branch ? 3 : 2;
    const int obase = branch * 128 + head * 32;

    const size_t pixBase = ((size_t)b * 16384 + hw) * 768;
    const float q = qkv[pixBase + obase + lane] * 0.17677669529663687f; // /sqrt(32)

    float scores[9];
    bool  valid[9];
#pragma unroll
    for (int ki = 0; ki < 3; ++ki) {
#pragma unroll
        for (int kj = 0; kj < 3; ++kj) {
            const int idx = ki * 3 + kj;
            const int hh = h + (ki - 1) * dil;
            const int ww = w + (kj - 1) * dil;
            const bool ok = (hh >= 0) & (hh < 128) & (ww >= 0) & (ww < 128);
            valid[idx] = ok;
            float s = 0.f;
            if (ok) { // wave-uniform branch
                const float* kp =
                    qkv + ((size_t)b * 16384 + hh * 128 + ww) * 768 + 256 + obase;
                float prod = q * kp[lane];
#pragma unroll
                for (int off = 16; off; off >>= 1)
                    prod += __shfl_xor(prod, off, 32);
                s = prod;
            }
            scores[idx] = s;
        }
    }

    float mx = scores[0];
#pragma unroll
    for (int i = 1; i < 9; ++i) mx = fmaxf(mx, scores[i]);
    float e[9], denom = 0.f;
#pragma unroll
    for (int i = 0; i < 9; ++i) { e[i] = __expf(scores[i] - mx); denom += e[i]; }
    const float inv = 1.f / denom;

    float out = 0.f;
#pragma unroll
    for (int ki = 0; ki < 3; ++ki) {
#pragma unroll
        for (int kj = 0; kj < 3; ++kj) {
            const int idx = ki * 3 + kj;
            if (valid[idx]) { // wave-uniform
                const int hh = h + (ki - 1) * dil;
                const int ww = w + (kj - 1) * dil;
                const float* vp =
                    qkv + ((size_t)b * 16384 + hh * 128 + ww) * 768 + 512 + obase;
                out = fmaf(e[idx] * inv, vp[lane], out);
            }
        }
    }

    xo[((size_t)b * 16384 + hw) * 256 + obase + lane] = out;
}

// ---------------------------------------------------------------------------
extern "C" void kernel_launch(void* const* d_in, const int* in_sizes, int n_in,
                              void* d_out, int out_size, void* d_ws, size_t ws_size,
                              hipStream_t stream) {
    (void)in_sizes; (void)n_in; (void)out_size; (void)ws_size;
    const float* x     = (const float*)d_in[0]; // [4,128,128,256]
    const float* Wqkv  = (const float*)d_in[1]; // [768,256]
    const float* Wproj = (const float*)d_in[2]; // [256,256]
    const float* bproj = (const float*)d_in[3]; // [256]
    float* out = (float*)d_out;                 // [4,128,128,256]

    float* qkv = (float*)d_ws;                  // 65536*768 f32 (201 MB)
    float* xo  = qkv + (size_t)65536 * 768;     // 65536*256 f32 ( 67 MB)

    const int M = 65536;
    dim3 blk(256);

    // Stage 1: qkv[m][o] = x[m][:] . Wqkv[o][:]
    dim3 g1(768 / 64, M / 128);
    gemm_bf16_wmma<256><<<g1, blk, 0, stream>>>(x, Wqkv, qkv, nullptr, M, 768);

    // Stage 2: attention (524288 waves, 8 per block)
    dilated_attn<<<dim3(65536), blk, 0, stream>>>(qkv, xo);

    // Stage 3: out[m][o] = xo[m][:] . Wproj[o][:] + b[o]
    dim3 g2(256 / 64, M / 128);
    gemm_bf16_wmma<256><<<g2, blk, 0, stream>>>(xo, Wproj, out, bproj, M, 256);
}
